// MoETransformer_52364241273507
// MI455X (gfx1250) — matrix operand order repaired and checked by reading, not compiled, not run on hardware
//
#include <hip/hip_runtime.h>
#include <hip/hip_bf16.h>

// ---------------- problem constants ----------------
#define S_   197
#define B_   64
#define D_   768
#define H_   12
#define HD_  64
#define T_   (S_ * B_)      // 12608 tokens = 197 M-tiles of 64
#define E_   8
#define DFF_ 3072
#define D3_  2304
#define SP_  224            // keys padded to 7*32 for WMMA K-steps

typedef __bf16 v16bf __attribute__((ext_vector_type(16)));
typedef float  v8f   __attribute__((ext_vector_type(8)));

union Frag { v16bf v; unsigned int u[8]; };
union AccU { v8f v; float f[8]; };

__device__ inline unsigned short f2bf(float f) {
    union { __bf16 b; unsigned short u; } cv;
    cv.b = (__bf16)f;
    return cv.u;
}
__device__ inline float bf2f(unsigned short u) {
    union { float f; unsigned int i; } cv;
    cv.i = ((unsigned int)u) << 16;
    return cv.f;
}
__device__ inline unsigned int pk2(float a, float b) {   // pack 2 bf16 into dword
    return (unsigned int)f2bf(a) | ((unsigned int)f2bf(b) << 16);
}

// A-matrix fragment (16x32 bf16), ISA 7.12.2: lane holds row M=lane&15.
// VGPR i<4: K = hi*8 + 2i; VGPR i>=4: K = 16 + hi*8 + 2(i-4); hi = lane>>4.
template<int LDP>
__device__ inline v16bf load_fragA(const unsigned short* tile, int rowBase, int kOff, int lane) {
    Frag f;
    const unsigned short* rp = tile + (rowBase + (lane & 15)) * LDP + kOff + (lane >> 4) * 8;
    f.u[0] = *(const unsigned int*)(rp + 0);
    f.u[1] = *(const unsigned int*)(rp + 2);
    f.u[2] = *(const unsigned int*)(rp + 4);
    f.u[3] = *(const unsigned int*)(rp + 6);
    f.u[4] = *(const unsigned int*)(rp + 16);
    f.u[5] = *(const unsigned int*)(rp + 18);
    f.u[6] = *(const unsigned int*)(rp + 20);
    f.u[7] = *(const unsigned int*)(rp + 22);
    return f.v;
}

// B-matrix fragment (32x16 bf16): lane holds column N=lane&15;
// same K order as A: VGPR i<4: K = hi*8 + 2i; VGPR i>=4: K = 16 + hi*8 + 2(i-4).
template<int LDP>
__device__ inline v16bf load_fragB(const unsigned short* tile, int rowBase, int kOff, int lane) {
    Frag f;
    const unsigned short* rp = tile + (rowBase + (lane & 15)) * LDP + kOff + (lane >> 4) * 8;
    #pragma unroll
    for (int i = 0; i < 8; ++i) f.u[i] = *(const unsigned int*)(rp + 2 * i + (i >= 4 ? 8 : 0));
    return f.v;
}

// ---------------- pipelined WMMA GEMM: Out[M,N] = act(A[M,K] * W[N,K]^T + bias) ----------------
// ACT: 0=none, 1=quickGELU.  GATHER: A row = gidx[row]>>1 (token id).
// SCATTER: out row = gidx[row] (token*2+slot).  RESID: outF += v.  OUTBF: bf16 out.
template<int ACT, bool GATHER, bool SCATTER, bool RESID, bool OUTBF>
__launch_bounds__(128)
__global__ void gemm64(const unsigned short* __restrict__ A, int lda,
                       const float* __restrict__ W,
                       const float* __restrict__ bias,
                       int K,
                       const int* __restrict__ gidx,
                       const int* __restrict__ nrowsPtr, int nrowsConst,
                       float* __restrict__ outF, unsigned short* __restrict__ outB,
                       int ldo)
{
    int nrows = nrowsPtr ? *nrowsPtr : nrowsConst;
    int rowBase = blockIdx.x * 64;
    if (rowBase >= nrows) return;
    int colBase = blockIdx.y * 64;

    // double-buffered tiles, stride 40 (80B rows: 16B-aligned b128 stores, conflict-free frags)
    __shared__ __align__(16) unsigned short As[2][64 * 40];
    __shared__ __align__(16) unsigned short Bs[2][64 * 40];

    int tid = threadIdx.x, lane = tid & 31, wave = tid >> 5;

    // each thread owns a contiguous 16-element (32B) segment of the A tile and of the W tile
    int ar = tid >> 1;            // row 0..63
    int ac = (tid & 1) * 16;      // col 0 or 16
    int gr = rowBase + ar;
    bool aval = (gr < nrows);
    long arow = aval ? (GATHER ? (long)(gidx[gr] >> 1) : (long)gr) : 0;
    const unsigned short* aptr = A + arow * lda + ac;
    const float*          wptr = W + (long)(colBase + ar) * K + ac;

    uint4  aReg0, aReg1;          // 16 bf16 (A)
    float4 wReg0, wReg1, wReg2, wReg3;   // 16 f32 (W)

    AccU acc[4];
    #pragma unroll
    for (int r = 0; r < 4; ++r)
        #pragma unroll
        for (int i = 0; i < 8; ++i) acc[r].f[i] = 0.f;

    auto loadTile = [&](int k0) {
        if (aval) {
            const uint4* p = (const uint4*)(aptr + k0);
            aReg0 = p[0]; aReg1 = p[1];
        } else {
            aReg0.x = aReg0.y = aReg0.z = aReg0.w = 0u;
            aReg1 = aReg0;
        }
        const float4* q = (const float4*)(wptr + k0);
        wReg0 = q[0]; wReg1 = q[1]; wReg2 = q[2]; wReg3 = q[3];
    };
    auto storeTile = [&](int buf) {
        *(uint4*)&As[buf][ar * 40 + ac]     = aReg0;
        *(uint4*)&As[buf][ar * 40 + ac + 8] = aReg1;
        uint4 b0, b1;
        b0.x = pk2(wReg0.x, wReg0.y); b0.y = pk2(wReg0.z, wReg0.w);
        b0.z = pk2(wReg1.x, wReg1.y); b0.w = pk2(wReg1.z, wReg1.w);
        b1.x = pk2(wReg2.x, wReg2.y); b1.y = pk2(wReg2.z, wReg2.w);
        b1.z = pk2(wReg3.x, wReg3.y); b1.w = pk2(wReg3.z, wReg3.w);
        *(uint4*)&Bs[buf][ar * 40 + ac]     = b0;
        *(uint4*)&Bs[buf][ar * 40 + ac + 8] = b1;
    };

    int nk = K >> 5;
    loadTile(0);
    storeTile(0);
    __syncthreads();

    for (int kt = 0; kt < nk; ++kt) {
        int buf = kt & 1;
        if (kt + 1 < nk) loadTile((kt + 1) << 5);            // prefetch next tile -> VGPRs
        if (kt + 2 < nk)                                     // deep prefetch of W (L2 hint)
            __builtin_prefetch(wptr + ((kt + 2) << 5), 0, 1);

        v16bf bfrag = load_fragB<40>(Bs[buf], wave * 16, 0, lane);
        #pragma unroll
        for (int rt = 0; rt < 4; ++rt) {
            v16bf afrag = load_fragA<40>(As[buf], rt * 16, 0, lane);
            acc[rt].v = __builtin_amdgcn_wmma_f32_16x16x32_bf16(
                false, afrag, false, bfrag, (short)0, acc[rt].v, false, false);
        }
        if (kt + 1 < nk) {
            storeTile(1 - buf);                              // fill other buffer
            __syncthreads();
        }
    }

    int n = colBase + wave * 16 + (lane & 15);
    int hi = lane >> 4;
    float bv = bias ? bias[n] : 0.f;
    #pragma unroll
    for (int rt = 0; rt < 4; ++rt) {
        #pragma unroll
        for (int vr = 0; vr < 8; ++vr) {
            int m = rowBase + rt * 16 + hi * 8 + vr;
            if (m < nrows) {
                float v = acc[rt].f[vr] + bv;
                if (ACT == 1) v = v * (1.f / (1.f + __expf(-1.702f * v)));  // quickGELU
                long orow = SCATTER ? (long)gidx[m] : (long)m;
                long off = orow * (long)ldo + n;
                if (RESID)       outF[off] += v;
                else if (OUTBF)  outB[off] = f2bf(v);
                else             outF[off] = v;
            }
        }
    }
}

// ---------------- fused attention: one block per (qtile of 32, head, batch) ----------------
__launch_bounds__(128)
__global__ void attn_kernel(const unsigned short* __restrict__ qkv,
                            unsigned short* __restrict__ o)
{
    int qt = blockIdx.x;   // 0..6
    int h  = blockIdx.y;   // 0..11
    int b  = blockIdx.z;   // 0..63

    __shared__ __align__(16) unsigned short Kl[SP_ * 72];  // [key][hd]
    __shared__ __align__(16) unsigned short Vt[64 * 232];  // [hd][key] (V transposed)
    __shared__ __align__(16) unsigned short Ql[32 * 72];   // [qrow][hd]
    __shared__ float          Sc[32 * SP_];                // scores f32
    __shared__ unsigned short Pl[32 * 232];                // probs bf16

    int tid = threadIdx.x, lane = tid & 31, wave = tid >> 5;

    // stage K (b128) and V (b128 load, transposed scalar store); zero-pad keys >= 197
    #pragma unroll
    for (int i = 0; i < 14; ++i) {
        int seg = tid + i * 128;             // 1792 segments: 224 keys x 8 segs of 8
        int key = seg >> 3, dch = (seg & 7) * 8;
        uint4 kv, vv;
        if (key < S_) {
            long base = ((long)(key * B_ + b)) * D3_ + h * HD_ + dch;
            kv = *(const uint4*)(qkv + base + D_);
            vv = *(const uint4*)(qkv + base + 2 * D_);
        } else {
            kv.x = kv.y = kv.z = kv.w = 0u;
            vv = kv;
        }
        *(uint4*)&Kl[key * 72 + dch] = kv;
        unsigned int uu[4] = { vv.x, vv.y, vv.z, vv.w };
        #pragma unroll
        for (int j = 0; j < 4; ++j) {
            Vt[(dch + 2 * j) * 232 + key]     = (unsigned short)(uu[j] & 0xffffu);
            Vt[(dch + 2 * j + 1) * 232 + key] = (unsigned short)(uu[j] >> 16);
        }
    }
    // stage Q tile (32 x 64, b128)
    #pragma unroll
    for (int i = 0; i < 2; ++i) {
        int seg = tid + i * 128;
        int r = seg >> 3, dch = (seg & 7) * 8;
        int s = qt * 32 + r;
        uint4 qv;
        if (s < S_) qv = *(const uint4*)(qkv + ((long)(s * B_ + b)) * D3_ + h * HD_ + dch);
        else { qv.x = qv.y = qv.z = qv.w = 0u; }
        *(uint4*)&Ql[r * 72 + dch] = qv;
    }
    __syncthreads();

    // scores = scale * Q K^T  (14 key-tiles across 4 waves)
    const float scale = 0.125f;   // 1/sqrt(64)
    for (int nt = wave; nt < 14; nt += 4) {
        AccU a0, a1;
        #pragma unroll
        for (int i = 0; i < 8; ++i) { a0.f[i] = 0.f; a1.f[i] = 0.f; }
        #pragma unroll
        for (int ks = 0; ks < 2; ++ks) {
            v16bf bf  = load_fragB<72>(Kl, nt * 16, ks * 32, lane);
            v16bf af0 = load_fragA<72>(Ql, 0,  ks * 32, lane);
            v16bf af1 = load_fragA<72>(Ql, 16, ks * 32, lane);
            a0.v = __builtin_amdgcn_wmma_f32_16x16x32_bf16(false, af0, false, bf, (short)0, a0.v, false, false);
            a1.v = __builtin_amdgcn_wmma_f32_16x16x32_bf16(false, af1, false, bf, (short)0, a1.v, false, false);
        }
        int hi = lane >> 4;
        int n = nt * 16 + (lane & 15);
        #pragma unroll
        for (int vr = 0; vr < 8; ++vr) {
            float s0 = a0.f[vr] * scale, s1 = a1.f[vr] * scale;
            if (n >= S_) { s0 = -1e30f; s1 = -1e30f; }   // mask padded keys
            Sc[(hi * 8 + vr) * SP_ + n]      = s0;
            Sc[(16 + hi * 8 + vr) * SP_ + n] = s1;
        }
    }
    __syncthreads();

    // row softmax (32 rows, one thread each), probs -> bf16
    if (tid < 32) {
        float mx = -1e30f;
        for (int k = 0; k < SP_; ++k) mx = fmaxf(mx, Sc[tid * SP_ + k]);
        float sum = 0.f;
        for (int k = 0; k < SP_; ++k) {
            float e = __expf(Sc[tid * SP_ + k] - mx);
            Sc[tid * SP_ + k] = e;
            sum += e;
        }
        float inv = 1.f / sum;
        for (int k = 0; k < SP_; ++k) Pl[tid * 232 + k] = f2bf(Sc[tid * SP_ + k] * inv);
    }
    __syncthreads();

    // O = P V  (each wave owns one 16-wide hd tile)
    {
        int nt = wave;
        AccU a0, a1;
        #pragma unroll
        for (int i = 0; i < 8; ++i) { a0.f[i] = 0.f; a1.f[i] = 0.f; }
        #pragma unroll
        for (int ks = 0; ks < 7; ++ks) {
            v16bf bf  = load_fragB<232>(Vt, nt * 16, ks * 32, lane);
            v16bf af0 = load_fragA<232>(Pl, 0,  ks * 32, lane);
            v16bf af1 = load_fragA<232>(Pl, 16, ks * 32, lane);
            a0.v = __builtin_amdgcn_wmma_f32_16x16x32_bf16(false, af0, false, bf, (short)0, a0.v, false, false);
            a1.v = __builtin_amdgcn_wmma_f32_16x16x32_bf16(false, af1, false, bf, (short)0, a1.v, false, false);
        }
        int hi = lane >> 4;
        int n = nt * 16 + (lane & 15);
        #pragma unroll
        for (int vr = 0; vr < 8; ++vr) {
            int m0 = qt * 32 + hi * 8 + vr;
            int m1 = m0 + 16;
            if (m0 < S_) o[((long)(m0 * B_ + b)) * D_ + h * HD_ + n] = f2bf(a0.f[vr]);
            if (m1 < S_) o[((long)(m1 * B_ + b)) * D_ + h * HD_ + n] = f2bf(a1.f[vr]);
        }
    }
}

// ---------------- LayerNorm: one block (256 thr) per token, f32 in -> bf16 out ----------------
__launch_bounds__(256)
__global__ void ln_kernel(const float* __restrict__ x,
                          const float* __restrict__ sc,
                          const float* __restrict__ bi,
                          unsigned short* __restrict__ out)
{
    int t = blockIdx.x;
    __shared__ float r1[256], r2[256];
    const float* xp = x + (long)t * D_;
    float v0 = xp[threadIdx.x], v1 = xp[threadIdx.x + 256], v2 = xp[threadIdx.x + 512];
    r1[threadIdx.x] = v0 + v1 + v2;
    r2[threadIdx.x] = v0 * v0 + v1 * v1 + v2 * v2;
    __syncthreads();
    for (int off = 128; off > 0; off >>= 1) {
        if (threadIdx.x < off) {
            r1[threadIdx.x] += r1[threadIdx.x + off];
            r2[threadIdx.x] += r2[threadIdx.x + off];
        }
        __syncthreads();
    }
    float mu = r1[0] * (1.f / D_);
    float var = r2[0] * (1.f / D_) - mu * mu;
    float rs = rsqrtf(var + 1e-5f);
    unsigned short* op = out + (long)t * D_;
    op[threadIdx.x]       = f2bf((v0 - mu) * rs * sc[threadIdx.x]       + bi[threadIdx.x]);
    op[threadIdx.x + 256] = f2bf((v1 - mu) * rs * sc[threadIdx.x + 256] + bi[threadIdx.x + 256]);
    op[threadIdx.x + 512] = f2bf((v2 - mu) * rs * sc[threadIdx.x + 512] + bi[threadIdx.x + 512]);
}

// ---------------- router: one wave32 per token (coalesced dot products + shfl reduce) --------
__launch_bounds__(256)
__global__ void route_kernel(const unsigned short* __restrict__ ln,
                             const float* __restrict__ gw,      // [E, D] (layer-0 gate)
                             float* __restrict__ logits,        // [T, E] -> d_out tail
                             float* __restrict__ wts,           // [T, 2]
                             int* __restrict__ lists,           // [E, T] entries = t*2+slot
                             int* __restrict__ counts)          // [E]
{
    int t    = (blockIdx.x * 256 + threadIdx.x) >> 5;  // token = global wave id (exact: 1576*8)
    int lane = threadIdx.x & 31;
    const unsigned short* a = ln + (long)t * D_;

    float p[E_];
    #pragma unroll
    for (int e = 0; e < E_; ++e) {
        float s = 0.f;
        const float* g = gw + e * D_;
        for (int k = lane; k < D_; k += 32) s += bf2f(a[k]) * g[k];
        #pragma unroll
        for (int m = 16; m; m >>= 1) s += __shfl_xor(s, m, 32);
        p[e] = s;
    }
    if (lane == 0) {
        #pragma unroll
        for (int e = 0; e < E_; ++e) logits[(long)t * E_ + e] = p[e];
        float mx = p[0];
        #pragma unroll
        for (int e = 1; e < E_; ++e) mx = fmaxf(mx, p[e]);
        float sum = 0.f;
        #pragma unroll
        for (int e = 0; e < E_; ++e) { p[e] = __expf(p[e] - mx); sum += p[e]; }
        float inv = 1.f / sum;
        #pragma unroll
        for (int e = 0; e < E_; ++e) p[e] *= inv;
        int i0 = 0;
        #pragma unroll
        for (int e = 1; e < E_; ++e) if (p[e] > p[i0]) i0 = e;
        int i1 = -1;
        #pragma unroll
        for (int e = 0; e < E_; ++e) if (e != i0 && (i1 < 0 || p[e] > p[i1])) i1 = e;
        float denom = p[i0] + p[i1];
        wts[2 * t]     = p[i0] / denom;
        wts[2 * t + 1] = p[i1] / denom;
        int pos0 = atomicAdd(&counts[i0], 1);
        lists[i0 * T_ + pos0] = t * 2;
        int pos1 = atomicAdd(&counts[i1], 1);
        lists[i1 * T_ + pos1] = t * 2 + 1;
    }
}

// x[t,d] += w0*out_g[2t,d] + w1*out_g[2t+1,d]  (fixed slot order -> deterministic)
__launch_bounds__(256)
__global__ void combine_kernel(float* __restrict__ x,
                               const unsigned short* __restrict__ og,
                               const float* __restrict__ wts)
{
    int idx = blockIdx.x * 256 + threadIdx.x;          // over T_*D_/4 float4s
    if (idx >= T_ * (D_ / 4)) return;
    int t  = idx / (D_ / 4);
    int d4 = idx - t * (D_ / 4);
    float* xp = x + (long)t * D_ + d4 * 4;
    float4 xv = *(float4*)xp;
    uint2 o0 = *(const uint2*)(og + (long)(2 * t) * D_ + d4 * 4);
    uint2 o1 = *(const uint2*)(og + (long)(2 * t + 1) * D_ + d4 * 4);
    float w0 = wts[2 * t], w1 = wts[2 * t + 1];
    xv.x += w0 * bf2f((unsigned short)(o0.x & 0xffffu)) + w1 * bf2f((unsigned short)(o1.x & 0xffffu));
    xv.y += w0 * bf2f((unsigned short)(o0.x >> 16))     + w1 * bf2f((unsigned short)(o1.x >> 16));
    xv.z += w0 * bf2f((unsigned short)(o0.y & 0xffffu)) + w1 * bf2f((unsigned short)(o1.y & 0xffffu));
    xv.w += w0 * bf2f((unsigned short)(o0.y >> 16))     + w1 * bf2f((unsigned short)(o1.y >> 16));
    *(float4*)xp = xv;
}

__global__ void zero_kernel(int* __restrict__ p, int n) {
    int i = threadIdx.x;
    if (i < n) p[i] = 0;
}

// ---------------- host orchestration ----------------
extern "C" void kernel_launch(void* const* d_in, const int* in_sizes, int n_in,
                              void* d_out, int out_size, void* d_ws, size_t ws_size,
                              hipStream_t stream)
{
    (void)in_sizes; (void)n_in; (void)out_size; (void)ws_size;

    const float* x_in       = (const float*)d_in[0];
    const float* moe_ln1_s  = (const float*)d_in[1];
    const float* moe_ln1_b  = (const float*)d_in[2];
    const float* moe_in_w   = (const float*)d_in[3];
    const float* moe_in_b   = (const float*)d_in[4];
    const float* moe_out_w  = (const float*)d_in[5];
    const float* moe_out_b  = (const float*)d_in[6];
    const float* moe_ln2_s  = (const float*)d_in[7];
    const float* moe_ln2_b  = (const float*)d_in[8];
    const float* moe_gate_w = (const float*)d_in[9];
    const float* moe_w1     = (const float*)d_in[10];
    const float* moe_b1     = (const float*)d_in[11];
    const float* moe_w2     = (const float*)d_in[12];
    const float* moe_b2     = (const float*)d_in[13];
    const float* std_ln1_s  = (const float*)d_in[14];
    const float* std_ln1_b  = (const float*)d_in[15];
    const float* std_in_w   = (const float*)d_in[16];
    const float* std_in_b   = (const float*)d_in[17];
    const float* std_out_w  = (const float*)d_in[18];
    const float* std_out_b  = (const float*)d_in[19];
    const float* std_ln2_s  = (const float*)d_in[20];
    const float* std_ln2_b  = (const float*)d_in[21];
    const float* std_fc_w   = (const float*)d_in[22];
    const float* std_fc_b   = (const float*)d_in[23];
    const float* std_proj_w = (const float*)d_in[24];
    const float* std_proj_b = (const float*)d_in[25];

    float* x      = (float*)d_out;              // residual stream [T, D]
    float* logits = x + (long)T_ * D_;          // first_logits [T, E]

    char* ws = (char*)d_ws;
    size_t off = 0;
    auto alloc = [&](size_t bytes) -> void* {
        void* p = ws + off;
        off = (off + bytes + 255) & ~(size_t)255;
        return p;
    };
    unsigned short* lnb   = (unsigned short*)alloc((size_t)T_ * D_ * 2);       // LN output (bf16)
    unsigned short* qkvB  = (unsigned short*)alloc((size_t)T_ * D3_ * 2);      // qkv (bf16)
    unsigned short* attno = (unsigned short*)alloc((size_t)T_ * D_ * 2);       // attention out (bf16)
    unsigned short* hbuf  = (unsigned short*)alloc((size_t)T_ * DFF_ * 2);     // FFN hidden (bf16)
    unsigned short* outg  = (unsigned short*)alloc((size_t)T_ * 2 * D_ * 2);   // [T,2,D] expert outs
    float* wts    = (float*)alloc((size_t)T_ * 2 * 4);
    int*   lists  = (int*)alloc((size_t)E_ * T_ * 4);
    int*   counts = (int*)alloc(256);

    hipMemcpyAsync(x, x_in, (size_t)T_ * D_ * sizeof(float),
                   hipMemcpyDeviceToDevice, stream);

    const int MT = T_ / 64;   // 197
    dim3 blk(128);

    // -------- MoE layers --------
    for (int l = 0; l < 2; ++l) {
        ln_kernel<<<T_, 256, 0, stream>>>(x, moe_ln1_s + l * D_, moe_ln1_b + l * D_, lnb);
        gemm64<0, false, false, false, true><<<dim3(MT, D3_ / 64), blk, 0, stream>>>(
            lnb, D_, moe_in_w + (long)l * D3_ * D_, moe_in_b + l * D3_, D_,
            nullptr, nullptr, T_, nullptr, qkvB, D3_);
        attn_kernel<<<dim3(7, H_, B_), blk, 0, stream>>>(qkvB, attno);
        gemm64<0, false, false, true, false><<<dim3(MT, D_ / 64), blk, 0, stream>>>(
            attno, D_, moe_out_w + (long)l * D_ * D_, moe_out_b + l * D_, D_,
            nullptr, nullptr, T_, x, nullptr, D_);
        ln_kernel<<<T_, 256, 0, stream>>>(x, moe_ln2_s + l * D_, moe_ln2_b + l * D_, lnb);
        if (l == 0) {
            zero_kernel<<<1, 32, 0, stream>>>(counts, E_);
            route_kernel<<<T_ / 8, 256, 0, stream>>>(
                lnb, moe_gate_w /* layer 0 */, logits, wts, lists, counts);
        }
        for (int e = 0; e < E_; ++e) {
            const float* w1 = moe_w1 + (long)(l * E_ + e) * DFF_ * D_;
            const float* b1 = moe_b1 + (long)(l * E_ + e) * DFF_;
            const float* w2 = moe_w2 + (long)(l * E_ + e) * D_ * DFF_;
            const float* b2 = moe_b2 + (long)(l * E_ + e) * D_;
            // h = quickGELU(gather(ln) @ w1^T + b1)   [count_e, DFF]
            gemm64<1, true, false, false, true><<<dim3(MT, DFF_ / 64), blk, 0, stream>>>(
                lnb, D_, w1, b1, D_, lists + e * T_, counts + e, 0, nullptr, hbuf, DFF_);
            // out_g[t*2+slot] = h @ w2^T + b2         (scatter)
            gemm64<0, false, true, false, true><<<dim3(MT, D_ / 64), blk, 0, stream>>>(
                hbuf, DFF_, w2, b2, DFF_, lists + e * T_, counts + e, 0, nullptr, outg, D_);
        }
        combine_kernel<<<(T_ * (D_ / 4) + 255) / 256, 256, 0, stream>>>(x, outg, wts);
    }

    // -------- standard layers --------
    for (int l = 0; l < 2; ++l) {
        ln_kernel<<<T_, 256, 0, stream>>>(x, std_ln1_s + l * D_, std_ln1_b + l * D_, lnb);
        gemm64<0, false, false, false, true><<<dim3(MT, D3_ / 64), blk, 0, stream>>>(
            lnb, D_, std_in_w + (long)l * D3_ * D_, std_in_b + l * D3_, D_,
            nullptr, nullptr, T_, nullptr, qkvB, D3_);
        attn_kernel<<<dim3(7, H_, B_), blk, 0, stream>>>(qkvB, attno);
        gemm64<0, false, false, true, false><<<dim3(MT, D_ / 64), blk, 0, stream>>>(
            attno, D_, std_out_w + (long)l * D_ * D_, std_out_b + l * D_, D_,
            nullptr, nullptr, T_, x, nullptr, D_);
        ln_kernel<<<T_, 256, 0, stream>>>(x, std_ln2_s + l * D_, std_ln2_b + l * D_, lnb);
        gemm64<1, false, false, false, true><<<dim3(MT, DFF_ / 64), blk, 0, stream>>>(
            lnb, D_, std_fc_w + (long)l * DFF_ * D_, std_fc_b + l * DFF_, D_,
            nullptr, nullptr, T_, nullptr, hbuf, DFF_);
        gemm64<0, false, false, true, false><<<dim3(MT, D_ / 64), blk, 0, stream>>>(
            hbuf, DFF_, std_proj_w + (long)l * D_ * DFF_, std_proj_b + l * D_, DFF_,
            nullptr, nullptr, T_, x, nullptr, D_);
    }
}
